// GraphConvolution_3891240370711
// MI455X (gfx1250) — matrix operand order repaired and checked by reading, not compiled, not run on hardware
//
#include <hip/hip_runtime.h>
#include <hip/hip_bf16.h>

#define N_NODES  100000
#define E_EDGES  3200000
#define DIN      512
#define DOUT     256
#define DCOMB    512   // [W0 | W1] combined output columns

typedef __attribute__((ext_vector_type(16))) __bf16 v16bf;
typedef __attribute__((ext_vector_type(8)))  __bf16 v8bf;
typedef __attribute__((ext_vector_type(4)))  __bf16 v4bf;
typedef __attribute__((ext_vector_type(8)))  float  v8f;

#define LDS_STRIDE 40   // 32 k-elements + 8 pad (80B row pitch: 16B-aligned, conflict-spread)

// ---------------------------------------------------------------------------
// Kernel 1: build Wt[n][k] = bf16( W{0,1}[k][n'] ), transposed + converted.
// Column-major-by-n layout so the WMMA B fragment (fixed n, 16 consecutive k)
// is a single contiguous 32-byte load.
// ---------------------------------------------------------------------------
__global__ void convert_w_kernel(const float* __restrict__ W0,
                                 const float* __restrict__ W1,
                                 __bf16* __restrict__ Wt) {
    const int n = blockIdx.x;  // 0..511 combined column
    const float* src = (n < DOUT) ? (W0 + n) : (W1 + (n - DOUT));
    for (int k = threadIdx.x; k < DIN; k += blockDim.x) {
        Wt[(size_t)n * DIN + k] = (__bf16)src[(size_t)k * DOUT];
    }
}

// ---------------------------------------------------------------------------
// Kernel 2: zero the output accumulator (d_out is poisoned by the harness).
// ---------------------------------------------------------------------------
__global__ void zero_kernel(float4* __restrict__ out, int n4) {
    int i = blockIdx.x * blockDim.x + threadIdx.x;
    if (i < n4) out[i] = make_float4(0.f, 0.f, 0.f, 0.f);
}

// ---------------------------------------------------------------------------
// Kernel 3: h[N,512] = x[N,512] @ [W0|W1]  via v_wmma_f32_16x16x32_bf16.
// One block = 32-row stripe x all 512 columns, so x is read from HBM exactly
// once. A tile (32x32) is converted f32->bf16 cooperatively and staged in LDS;
// each of the 8 waves owns 16 rows x 128 cols (8 C tiles), so one A fragment
// feeds 8 WMMAs per k-step. All 8 B fragments are loaded into registers first
// (one load clause) so the WMMA burst drains them without per-tile waits.
// grid = (N/32).
// ---------------------------------------------------------------------------
__global__ void __launch_bounds__(256)
gemm_bf16_wmma_kernel(const float* __restrict__ x,
                      const __bf16* __restrict__ Wt,
                      float* __restrict__ h) {
    __shared__ __bf16 As[32 * LDS_STRIDE];

    const int tid  = threadIdx.x;
    const int lane = tid & 31;
    const int wave = tid >> 5;
    const int m0   = blockIdx.x * 32;

    // wave tile: LDS-local rows mrow0..mrow0+15, cols n0..n0+127
    const int mrow0 = (wave >> 2) * 16;   // 0 or 16
    const int n0    = (wave & 3) * 128;

    // cooperative A-load mapping: thread t loads one float4 of the 32x32 tile
    const int ldrow = tid >> 3;           // 0..31
    const int ldk   = (tid & 7) * 4;      // 0,4,...,28
    const float* xsrc = x + (size_t)(m0 + ldrow) * DIN + ldk;

    // per-lane A fragment position in LDS
    const int arow = mrow0 + (lane & 15);
    const int kbA  = (lane >> 4) * 16;    // 0 or 16
    const __bf16* afrag = &As[arow * LDS_STRIDE + kbA];

    // per-lane B fragment base (tile t adds 16 columns = 16*DIN elements)
    const int kbB = (lane >> 4) * 16;     // 0 or 16
    const __bf16* wbase = Wt + (size_t)(n0 + (lane & 15)) * DIN + kbB;

    v8f c[8] = {};

    float4 f = *(const float4*)xsrc;      // k-step 0 prefetch

    #pragma unroll 2
    for (int ks = 0; ks < DIN / 32; ++ks) {
        const int k0 = ks * 32;

        // stage converted tile into LDS
        v4bf bv;
        bv[0] = (__bf16)f.x; bv[1] = (__bf16)f.y;
        bv[2] = (__bf16)f.z; bv[3] = (__bf16)f.w;
        if (ks) __syncthreads();          // previous tile fully consumed
        *(v4bf*)(&As[ldrow * LDS_STRIDE + ldk]) = bv;
        __syncthreads();

        // read A fragment (two 16B LDS loads)
        v8bf alo = *(const v8bf*)(afrag);
        v8bf ahi = *(const v8bf*)(afrag + 8);
        v16bf a;
        #pragma unroll
        for (int i = 0; i < 8; ++i) { a[i] = alo[i]; a[8 + i] = ahi[i]; }

        // prefetch next k-step's global data while WMMAs run
        if (ks + 1 < DIN / 32) f = *(const float4*)(xsrc + (ks + 1) * 32);

        // pull all 8 B fragments into registers (one load clause) ...
        v16bf b[8];
        #pragma unroll
        for (int t = 0; t < 8; ++t)
            b[t] = *(const v16bf*)(wbase + (size_t)t * 16 * DIN + k0);

        // ... then drain them through the matrix pipe
        #pragma unroll
        for (int t = 0; t < 8; ++t) {
            // (neg_a, A, neg_b, B, c_mod, C, reuse_a, reuse_b)
            c[t] = __builtin_amdgcn_wmma_f32_16x16x32_bf16(false, a, false, b[t],
                                                           (short)0, c[t], false, false);
        }
    }

    // D layout: col = n0 + t*16 + (lane&15); VGPR r -> row +r + 8*(lane>>4)
    const int mrowD = m0 + mrow0 + 8 * (lane >> 4);
    #pragma unroll
    for (int t = 0; t < 8; ++t) {
        float* hp = h + (size_t)mrowD * DCOMB + n0 + t * 16 + (lane & 15);
        #pragma unroll
        for (int r = 0; r < 8; ++r) hp[(size_t)r * DCOMB] = c[t][r];
    }
}

// ---------------------------------------------------------------------------
// Kernel 4: COO SpMM with folded conv weight:
//   out[row[e], :] += conv_w[s] * val[e] * h[col[e], s*256 : s*256+256]
// Each wave batches 32 edges' metadata, broadcasts via __shfl; each lane owns
// 8 consecutive output floats (2 x float4 gather, 8 x f32 HW atomics).
// ---------------------------------------------------------------------------
__global__ void __launch_bounds__(256)
spmm_atomic_kernel(const int*   __restrict__ row,
                   const int*   __restrict__ col,
                   const float* __restrict__ val,
                   const float* __restrict__ h,
                   float*       __restrict__ out,
                   const float* __restrict__ conv_w,
                   int support) {
    const int lane   = threadIdx.x & 31;
    const int wid    = blockIdx.x * (blockDim.x >> 5) + (threadIdx.x >> 5);
    const int nwaves = gridDim.x * (blockDim.x >> 5);
    const float cw   = conv_w[support];
    const int  coff  = support * DOUT;

    for (int base = wid * 32; base < E_EDGES; base += nwaves * 32) {
        const int e = base + lane;
        int r = 0, ci = 0; float v = 0.f;
        if (e < E_EDGES) { r = row[e]; ci = col[e]; v = val[e]; }
        const int cnt = min(32, E_EDGES - base);

        for (int j = 0; j < cnt; ++j) {
            const int   rj = __shfl(r,  j);
            const int   cj = __shfl(ci, j);
            const float vj = __shfl(v,  j) * cw;

            // prefetch next edge's source row into cache
            if (j + 1 < cnt) {
                const int cn = __shfl(ci, j + 1);
                __builtin_prefetch(h + (size_t)cn * DCOMB + coff + lane * 8, 0, 0);
            }

            const float4* hp = (const float4*)(h + (size_t)cj * DCOMB + coff) + lane * 2;
            float4 a = hp[0];
            float4 b = hp[1];

            float* o = out + (size_t)rj * DOUT + lane * 8;
            unsafeAtomicAdd(o + 0, a.x * vj);
            unsafeAtomicAdd(o + 1, a.y * vj);
            unsafeAtomicAdd(o + 2, a.z * vj);
            unsafeAtomicAdd(o + 3, a.w * vj);
            unsafeAtomicAdd(o + 4, b.x * vj);
            unsafeAtomicAdd(o + 5, b.y * vj);
            unsafeAtomicAdd(o + 6, b.z * vj);
            unsafeAtomicAdd(o + 7, b.w * vj);
        }
    }
}

// ---------------------------------------------------------------------------
// Kernel 5: in-place ReLU on the accumulated output.
// ---------------------------------------------------------------------------
__global__ void relu_kernel(float4* __restrict__ out, int n4) {
    int i = blockIdx.x * blockDim.x + threadIdx.x;
    if (i < n4) {
        float4 v = out[i];
        v.x = fmaxf(v.x, 0.f);
        v.y = fmaxf(v.y, 0.f);
        v.z = fmaxf(v.z, 0.f);
        v.w = fmaxf(v.w, 0.f);
        out[i] = v;
    }
}

// ---------------------------------------------------------------------------
// Launch: inputs are (x, row0, col0, val0, row1, col1, val1, W0, W1, conv_w).
// ws layout: [ Wt bf16 512*512 | h f32 N*512 ]  (~205.3 MB)
// ---------------------------------------------------------------------------
extern "C" void kernel_launch(void* const* d_in, const int* in_sizes, int n_in,
                              void* d_out, int out_size, void* d_ws, size_t ws_size,
                              hipStream_t stream) {
    const float* x      = (const float*)d_in[0];
    const int*   row0   = (const int*)  d_in[1];
    const int*   col0   = (const int*)  d_in[2];
    const float* val0   = (const float*)d_in[3];
    const int*   row1   = (const int*)  d_in[4];
    const int*   col1   = (const int*)  d_in[5];
    const float* val1   = (const float*)d_in[6];
    const float* W0     = (const float*)d_in[7];
    const float* W1     = (const float*)d_in[8];
    const float* conv_w = (const float*)d_in[9];

    char*   ws  = (char*)d_ws;
    __bf16* Wt  = (__bf16*)ws;
    float*  h   = (float*)(ws + (size_t)DCOMB * DIN * sizeof(__bf16));
    float*  out = (float*)d_out;

    // 1. transpose + f32->bf16 convert weights
    convert_w_kernel<<<DCOMB, 256, 0, stream>>>(W0, W1, Wt);

    // 2. zero output accumulator
    const int n4 = (N_NODES * DOUT) / 4;
    zero_kernel<<<(n4 + 255) / 256, 256, 0, stream>>>((float4*)out, n4);

    // 3. dense GEMM: h = x @ [W0|W1]  (bf16 WMMA, f32 accumulate, LDS-staged A)
    gemm_bf16_wmma_kernel<<<N_NODES / 32, 256, 0, stream>>>(x, Wt, h);

    // 4. two SpMM passes with conv weights folded into the atomic accumulate
    spmm_atomic_kernel<<<4096, 256, 0, stream>>>(row0, col0, val0, h, out, conv_w, 0);
    spmm_atomic_kernel<<<4096, 256, 0, stream>>>(row1, col1, val1, h, out, conv_w, 1);

    // 5. relu in place
    relu_kernel<<<(n4 + 255) / 256, 256, 0, stream>>>((float4*)out, n4);
}
